// SelectiveWKV_1B_21122649162004
// MI455X (gfx1250) — matrix operand order repaired and checked by reading, not compiled, not run on hardware
//
#include <hip/hip_runtime.h>

#define D_MODEL 2048
#define HEAD    64
#define NHEAD   32
#define BATCH   2
#define SEQ     1024
#define ROWS    (BATCH * SEQ)   // GEMM M = 2048

typedef __bf16 bf16;
typedef __attribute__((ext_vector_type(16))) __bf16 v16bf;
typedef __attribute__((ext_vector_type(8)))  float  v8f;

union FragAB {
    v16bf v;
    uint4 u[2];
};

#define LD4(p) (*(const uint4*)(p))

__device__ __forceinline__ float sigmoidf_(float x) {
    return 1.0f / (1.0f + __expf(-x));
}

// split fp32 into bf16 hi + bf16 residual (bf16x3 emulated-fp32 GEMM inputs)
__device__ __forceinline__ void bf16split(float x, bf16* hi, bf16* lo) {
    bf16 h = (bf16)x;
    *hi = h;
    *lo = (bf16)(x - (float)h);
}

// ---------------------------------------------------------------------------
// LayerNorm over rows of [ROWS, D_MODEL]; writes xn as bf16 hi/lo split.
// ---------------------------------------------------------------------------
__global__ void ln_split_kernel(const float* __restrict__ x,
                                const float* __restrict__ gamma,
                                const float* __restrict__ beta,
                                bf16* __restrict__ xnHi,
                                bf16* __restrict__ xnLo) {
    __shared__ float red[256];
    const int row = blockIdx.x;
    const int tid = threadIdx.x;
    const float* xr = x + (size_t)row * D_MODEL;

    float v[8];
    float s = 0.f, s2 = 0.f;
#pragma unroll
    for (int i = 0; i < 8; ++i) {
        float a = xr[tid + i * 256];
        v[i] = a;
        s += a;
        s2 += a * a;
    }
    red[tid] = s;
    __syncthreads();
    for (int off = 128; off; off >>= 1) {
        if (tid < off) red[tid] += red[tid + off];
        __syncthreads();
    }
    const float mean = red[0] * (1.0f / D_MODEL);
    __syncthreads();
    red[tid] = s2;
    __syncthreads();
    for (int off = 128; off; off >>= 1) {
        if (tid < off) red[tid] += red[tid + off];
        __syncthreads();
    }
    const float var  = red[0] * (1.0f / D_MODEL) - mean * mean;
    const float rstd = rsqrtf(var + 1e-5f);

#pragma unroll
    for (int i = 0; i < 8; ++i) {
        int c = tid + i * 256;
        float xn = (v[i] - mean) * rstd * gamma[c] + beta[c];
        bf16 h, l;
        bf16split(xn, &h, &l);
        xnHi[(size_t)row * D_MODEL + c] = h;
        xnLo[(size_t)row * D_MODEL + c] = l;
    }
}

// ---------------------------------------------------------------------------
// Transpose W[K,N] -> Wt[N,K] with bf16 hi/lo split (WMMA B operand wants
// contiguous-along-K storage so fragment loads are global_load_b128).
// ---------------------------------------------------------------------------
__global__ void wsplitT_kernel(const float* __restrict__ W,   // [K, N] = [2048,2048]
                               bf16* __restrict__ WtHi,       // [N, K]
                               bf16* __restrict__ WtLo) {
    __shared__ float tile[32][33];
    const int kT = blockIdx.y * 32;
    const int nT = blockIdx.x * 32;
    const int tx = threadIdx.x;   // 0..31
    const int ty = threadIdx.y;   // 0..7
#pragma unroll
    for (int i = 0; i < 4; ++i)
        tile[ty + i * 8][tx] = W[(size_t)(kT + ty + i * 8) * D_MODEL + nT + tx];
    __syncthreads();
#pragma unroll
    for (int i = 0; i < 4; ++i) {
        float val = tile[tx][ty + i * 8];        // = W[kT+tx][nT+ty+i*8]
        int nrow = nT + ty + i * 8;
        int kcol = kT + tx;
        bf16 h, l;
        bf16split(val, &h, &l);
        WtHi[(size_t)nrow * D_MODEL + kcol] = h;
        WtLo[(size_t)nrow * D_MODEL + kcol] = l;
    }
}

// ---------------------------------------------------------------------------
// Epilogue helper: C/D layout puts VGPR g at row = row0 + g, col fixed/lane.
// ---------------------------------------------------------------------------
template <int EPI>
__device__ __forceinline__ void emit_tile(const v8f& acc, int row0, int col,
                                          const float* __restrict__ bias,
                                          float* __restrict__ outF,
                                          bf16* __restrict__ outHi,
                                          bf16* __restrict__ outLo) {
#pragma unroll
    for (int g = 0; g < 8; ++g) {
        int row = row0 + g;
        float vv = acc[g];
        if (EPI == 2) vv = sigmoidf_(vv + bias[col]);
        if (EPI == 3) vv = sigmoidf_(vv);
        if (EPI == 1) {
            bf16 hh, ll;
            bf16split(vv, &hh, &ll);
            outHi[(size_t)row * D_MODEL + col] = hh;
            outLo[(size_t)row * D_MODEL + col] = ll;
        } else {
            outF[(size_t)row * D_MODEL + col] = vv;
        }
    }
}

// ---------------------------------------------------------------------------
// bf16x3 emulated-fp32 GEMM: C[M,N] = A[M,K] * B^T (B stored [N,K]).
// Per-wave 32x64 tile (2 m-tiles x 4 n-tiles = 8 accumulators); 8 waves/WG
// arranged 2(m) x 4(n) -> 64x256 per workgroup.
// Addressing: uniform SGPR base + one 32-bit lane offset per matrix side;
// all tile deltas (16*K elements = 64KB, ...) are compile-time immediates
// that fold into the 24-bit instruction offset -> no per-tile pointers.
// 3 WMMAs per K=32 per tile: Ahi*Bhi + Ahi*Blo + Alo*Bhi (fp32 acc).
// ---------------------------------------------------------------------------
template <int EPI>
__global__ void __launch_bounds__(256, 1)
gemm_bf16x3_kernel(const bf16* __restrict__ Ahi, const bf16* __restrict__ Alo,
                   const bf16* __restrict__ Bhi, const bf16* __restrict__ Blo,
                   const float* __restrict__ bias,
                   float* __restrict__ outF,
                   bf16* __restrict__ outHi, bf16* __restrict__ outLo) {
    const int K = D_MODEL;
    const int lane = threadIdx.x & 31;
    const int wave = threadIdx.x >> 5;          // wave32 on gfx1250
    const int l15  = lane & 15;
    const int h    = lane >> 4;                 // lane half-group
    const int m0   = blockIdx.y * 64  + (wave & 1) * 32;
    const int n0   = blockIdx.x * 256 + (wave >> 1) * 64;

    // lane-varying element offsets (shared by hi and lo arrays)
    const int offA = (m0 + l15) * K + h * 16;   // A: chunks at +0 and +8
    const int offB = (n0 + l15) * K + h * 16;   // B: chunks at +0 and +8

    v8f c00 = {}, c01 = {}, c02 = {}, c03 = {};
    v8f c10 = {}, c11 = {}, c12 = {}, c13 = {};

    for (int k0 = 0; k0 < K; k0 += 32) {
        const int ia = offA + k0;
        const int ib = offB + k0;

        FragAB A0H, A1H, A0L, A1L;
        FragAB B0H, B1H, B2H, B3H, B0L, B1L, B2L, B3L;

        A0H.u[0] = LD4(Ahi + ia);              A0H.u[1] = LD4(Ahi + ia + 8);
        A1H.u[0] = LD4(Ahi + ia + 16 * K);     A1H.u[1] = LD4(Ahi + ia + 16 * K + 8);
        A0L.u[0] = LD4(Alo + ia);              A0L.u[1] = LD4(Alo + ia + 8);
        A1L.u[0] = LD4(Alo + ia + 16 * K);     A1L.u[1] = LD4(Alo + ia + 16 * K + 8);

        B0H.u[0] = LD4(Bhi + ib);              B0H.u[1] = LD4(Bhi + ib + 8);
        B1H.u[0] = LD4(Bhi + ib + 16 * K);     B1H.u[1] = LD4(Bhi + ib + 16 * K + 8);
        B2H.u[0] = LD4(Bhi + ib + 32 * K);     B2H.u[1] = LD4(Bhi + ib + 32 * K + 8);
        B3H.u[0] = LD4(Bhi + ib + 48 * K);     B3H.u[1] = LD4(Bhi + ib + 48 * K + 8);
        B0L.u[0] = LD4(Blo + ib);              B0L.u[1] = LD4(Blo + ib + 8);
        B1L.u[0] = LD4(Blo + ib + 16 * K);     B1L.u[1] = LD4(Blo + ib + 16 * K + 8);
        B2L.u[0] = LD4(Blo + ib + 32 * K);     B2L.u[1] = LD4(Blo + ib + 32 * K + 8);
        B3L.u[0] = LD4(Blo + ib + 48 * K);     B3L.u[1] = LD4(Blo + ib + 48 * K + 8);

        // stream the next K-slab (global_prefetch_b8; speculative, OOB-safe)
        __builtin_prefetch(Ahi + ia + 32, 0, 1);
        __builtin_prefetch(Bhi + ib + 32, 0, 1);

        c00 = __builtin_amdgcn_wmma_f32_16x16x32_bf16(false, A0H.v, false, B0H.v, (short)0, c00, false, false);
        c00 = __builtin_amdgcn_wmma_f32_16x16x32_bf16(false, A0H.v, false, B0L.v, (short)0, c00, false, false);
        c00 = __builtin_amdgcn_wmma_f32_16x16x32_bf16(false, A0L.v, false, B0H.v, (short)0, c00, false, false);

        c01 = __builtin_amdgcn_wmma_f32_16x16x32_bf16(false, A0H.v, false, B1H.v, (short)0, c01, false, false);
        c01 = __builtin_amdgcn_wmma_f32_16x16x32_bf16(false, A0H.v, false, B1L.v, (short)0, c01, false, false);
        c01 = __builtin_amdgcn_wmma_f32_16x16x32_bf16(false, A0L.v, false, B1H.v, (short)0, c01, false, false);

        c02 = __builtin_amdgcn_wmma_f32_16x16x32_bf16(false, A0H.v, false, B2H.v, (short)0, c02, false, false);
        c02 = __builtin_amdgcn_wmma_f32_16x16x32_bf16(false, A0H.v, false, B2L.v, (short)0, c02, false, false);
        c02 = __builtin_amdgcn_wmma_f32_16x16x32_bf16(false, A0L.v, false, B2H.v, (short)0, c02, false, false);

        c03 = __builtin_amdgcn_wmma_f32_16x16x32_bf16(false, A0H.v, false, B3H.v, (short)0, c03, false, false);
        c03 = __builtin_amdgcn_wmma_f32_16x16x32_bf16(false, A0H.v, false, B3L.v, (short)0, c03, false, false);
        c03 = __builtin_amdgcn_wmma_f32_16x16x32_bf16(false, A0L.v, false, B3H.v, (short)0, c03, false, false);

        c10 = __builtin_amdgcn_wmma_f32_16x16x32_bf16(false, A1H.v, false, B0H.v, (short)0, c10, false, false);
        c10 = __builtin_amdgcn_wmma_f32_16x16x32_bf16(false, A1H.v, false, B0L.v, (short)0, c10, false, false);
        c10 = __builtin_amdgcn_wmma_f32_16x16x32_bf16(false, A1L.v, false, B0H.v, (short)0, c10, false, false);

        c11 = __builtin_amdgcn_wmma_f32_16x16x32_bf16(false, A1H.v, false, B1H.v, (short)0, c11, false, false);
        c11 = __builtin_amdgcn_wmma_f32_16x16x32_bf16(false, A1H.v, false, B1L.v, (short)0, c11, false, false);
        c11 = __builtin_amdgcn_wmma_f32_16x16x32_bf16(false, A1L.v, false, B1H.v, (short)0, c11, false, false);

        c12 = __builtin_amdgcn_wmma_f32_16x16x32_bf16(false, A1H.v, false, B2H.v, (short)0, c12, false, false);
        c12 = __builtin_amdgcn_wmma_f32_16x16x32_bf16(false, A1H.v, false, B2L.v, (short)0, c12, false, false);
        c12 = __builtin_amdgcn_wmma_f32_16x16x32_bf16(false, A1L.v, false, B2H.v, (short)0, c12, false, false);

        c13 = __builtin_amdgcn_wmma_f32_16x16x32_bf16(false, A1H.v, false, B3H.v, (short)0, c13, false, false);
        c13 = __builtin_amdgcn_wmma_f32_16x16x32_bf16(false, A1H.v, false, B3L.v, (short)0, c13, false, false);
        c13 = __builtin_amdgcn_wmma_f32_16x16x32_bf16(false, A1L.v, false, B3H.v, (short)0, c13, false, false);
    }

    const int r0 = m0 + h * 8;
    const int r1 = m0 + 16 + h * 8;
    const int cB = n0 + l15;
    emit_tile<EPI>(c00, r0, cB,      bias, outF, outHi, outLo);
    emit_tile<EPI>(c01, r0, cB + 16, bias, outF, outHi, outLo);
    emit_tile<EPI>(c02, r0, cB + 32, bias, outF, outHi, outLo);
    emit_tile<EPI>(c03, r0, cB + 48, bias, outF, outHi, outLo);
    emit_tile<EPI>(c10, r1, cB,      bias, outF, outHi, outLo);
    emit_tile<EPI>(c11, r1, cB + 16, bias, outF, outHi, outLo);
    emit_tile<EPI>(c12, r1, cB + 32, bias, outF, outHi, outLo);
    emit_tile<EPI>(c13, r1, cB + 48, bias, outF, outHi, outLo);
}

// ---------------------------------------------------------------------------
// Sequential WKV scan. One block per (b,h); 256 threads: thread = (ib, j),
// owns state rows [ib*16, ib*16+16) of column j in registers.
//   state = (1-w_t) (x) row-broadcast * state + k_t (outer) v_t
//   y_t   = r_t . state   (column-wise dot, reduced through LDS)
// y written as bf16 hi/lo for the final WMMA GEMM; final state -> stateOut.
// ---------------------------------------------------------------------------
__global__ void wkv_scan_kernel(const float* __restrict__ kB, const float* __restrict__ vB,
                                const float* __restrict__ wB, const float* __restrict__ rB,
                                bf16* __restrict__ yHi, bf16* __restrict__ yLo,
                                float* __restrict__ stateOut) {
    __shared__ float kbuf[64], vbuf[64], wbuf[64], rbuf[64], part[256];
    const int bh  = blockIdx.x;       // 0..63
    const int b   = bh >> 5;          // /NHEAD
    const int hd  = bh & 31;          // %NHEAD
    const int tid = threadIdx.x;
    const int j   = tid & 63;
    const int ib  = tid >> 6;         // 0..3
    const size_t chanBase = (size_t)hd * HEAD;

    float s[16];
#pragma unroll
    for (int u = 0; u < 16; ++u) s[u] = 0.f;

    for (int t = 0; t < SEQ; ++t) {
        const size_t idx = ((size_t)(b * SEQ + t)) * D_MODEL + chanBase;
        __syncthreads();   // previous iteration fully consumed LDS
        if (ib == 0)      kbuf[j] = kB[idx + j];
        else if (ib == 1) vbuf[j] = vB[idx + j];
        else if (ib == 2) wbuf[j] = wB[idx + j];
        else              rbuf[j] = rB[idx + j];
        __syncthreads();

        const float vj = vbuf[j];
        float pp = 0.f;
#pragma unroll
        for (int u = 0; u < 16; ++u) {
            int i = ib * 16 + u;
            s[u] = (1.0f - wbuf[i]) * s[u] + kbuf[i] * vj;
            pp += rbuf[i] * s[u];
        }
        part[tid] = pp;
        __syncthreads();
        if (ib == 0) {
            float y = part[j] + part[64 + j] + part[128 + j] + part[192 + j];
            bf16 hh, ll;
            bf16split(y, &hh, &ll);
            yHi[idx + j] = hh;
            yLo[idx + j] = ll;
        }
    }
    // final state [B,H,N,N]
#pragma unroll
    for (int u = 0; u < 16; ++u) {
        int i = ib * 16 + u;
        stateOut[(((size_t)b * NHEAD + hd) * HEAD + i) * HEAD + j] = s[u];
    }
}

// ---------------------------------------------------------------------------
extern "C" void kernel_launch(void* const* d_in, const int* in_sizes, int n_in,
                              void* d_out, int out_size, void* d_ws, size_t ws_size,
                              hipStream_t stream) {
    (void)in_sizes; (void)n_in; (void)out_size; (void)ws_size;

    const float* x     = (const float*)d_in[0];
    const float* Wx    = (const float*)d_in[1];
    const float* Ww    = (const float*)d_in[2];
    const float* bw    = (const float*)d_in[3];
    const float* Wk    = (const float*)d_in[4];
    const float* Wv    = (const float*)d_in[5];
    const float* Wr    = (const float*)d_in[6];
    const float* Wout  = (const float*)d_in[7];
    const float* gamma = (const float*)d_in[8];
    const float* beta  = (const float*)d_in[9];

    float* out      = (float*)d_out;                           // [B,T,C]
    float* stateOut = out + (size_t)BATCH * SEQ * D_MODEL;     // [B,H,N,N]

    char* ws = (char*)d_ws;
    const size_t HALF = (size_t)D_MODEL * ROWS * sizeof(bf16); // 8 MB per bf16 matrix
    bf16* xnHi  = (bf16*)(ws + 0 * HALF);
    bf16* xnLo  = (bf16*)(ws + 1 * HALF);
    bf16* wtHi  = (bf16*)(ws + 2 * HALF);   // reused per weight (stream-serialized)
    bf16* wtLo  = (bf16*)(ws + 3 * HALF);
    bf16* actHi = (bf16*)(ws + 4 * HALF);   // x1 first, then y (x1 dead after w-GEMM)
    bf16* actLo = (bf16*)(ws + 5 * HALF);
    float* kBuf = (float*)(ws + 6 * HALF);
    float* vBuf = kBuf + (size_t)ROWS * D_MODEL;
    float* wBuf = vBuf + (size_t)ROWS * D_MODEL;
    float* rBuf = wBuf + (size_t)ROWS * D_MODEL;

    const dim3 bLN(256), gLN(ROWS);
    const dim3 bT(32, 8), gT(D_MODEL / 32, D_MODEL / 32);
    const dim3 bG(256),  gG(D_MODEL / 256, ROWS / 64);

    // xn = LayerNorm(x) -> bf16 hi/lo
    ln_split_kernel<<<gLN, bLN, 0, stream>>>(x, gamma, beta, xnHi, xnLo);

    // x1 = xn @ Wx  (bf16 split out)
    wsplitT_kernel<<<gT, bT, 0, stream>>>(Wx, wtHi, wtLo);
    gemm_bf16x3_kernel<1><<<gG, bG, 0, stream>>>(xnHi, xnLo, wtHi, wtLo, nullptr,
                                                 nullptr, actHi, actLo);
    // w = sigmoid(x1 @ Ww + bw)
    wsplitT_kernel<<<gT, bT, 0, stream>>>(Ww, wtHi, wtLo);
    gemm_bf16x3_kernel<2><<<gG, bG, 0, stream>>>(actHi, actLo, wtHi, wtLo, bw,
                                                 wBuf, nullptr, nullptr);
    // k = xn @ Wk
    wsplitT_kernel<<<gT, bT, 0, stream>>>(Wk, wtHi, wtLo);
    gemm_bf16x3_kernel<0><<<gG, bG, 0, stream>>>(xnHi, xnLo, wtHi, wtLo, nullptr,
                                                 kBuf, nullptr, nullptr);
    // v = xn @ Wv
    wsplitT_kernel<<<gT, bT, 0, stream>>>(Wv, wtHi, wtLo);
    gemm_bf16x3_kernel<0><<<gG, bG, 0, stream>>>(xnHi, xnLo, wtHi, wtLo, nullptr,
                                                 vBuf, nullptr, nullptr);
    // r = sigmoid(xn @ Wr)
    wsplitT_kernel<<<gT, bT, 0, stream>>>(Wr, wtHi, wtLo);
    gemm_bf16x3_kernel<3><<<gG, bG, 0, stream>>>(xnHi, xnLo, wtHi, wtLo, nullptr,
                                                 rBuf, nullptr, nullptr);

    // recurrent scan: y (bf16 split into act slots), final state -> d_out tail
    wkv_scan_kernel<<<BATCH * NHEAD, 256, 0, stream>>>(kBuf, vBuf, wBuf, rBuf,
                                                       actHi, actLo, stateOut);

    // out = y @ Wout (fp32 -> d_out)
    wsplitT_kernel<<<gT, bT, 0, stream>>>(Wout, wtHi, wtLo);
    gemm_bf16x3_kernel<0><<<gG, bG, 0, stream>>>(actHi, actLo, wtHi, wtLo, nullptr,
                                                 out, nullptr, nullptr);
}
